// RelationProposalModel_77704548319690
// MI455X (gfx1250) — matrix operand order repaired and checked by reading, not compiled, not run on hardware
//
#include <hip/hip_runtime.h>
#include <cstdint>
#include <cstddef>

// ---------------------------------------------------------------------------
// Problem constants (from the reference)
// ---------------------------------------------------------------------------
#define NN    2048      // nodes
#define CC    151       // classes
#define EE    200       // sem dim
#define DROI  4096
#define GG    128       // pos dim
#define HH    512       // hidden
#define KREL  384       // 328 (=128+200) padded to multiple of 64

typedef __attribute__((ext_vector_type(16))) __bf16 v16bf;
typedef __attribute__((ext_vector_type(8)))  float  v8f;

union Frag16 { v16bf v; uint4 q[2]; };

// ---------------------------------------------------------------------------
// Elementwise helpers
// ---------------------------------------------------------------------------
__global__ void zero_f32_kernel(float* __restrict__ p, size_t n) {
    for (size_t i = blockIdx.x * (size_t)blockDim.x + threadIdx.x; i < n;
         i += (size_t)gridDim.x * blockDim.x)
        p[i] = 0.f;
}

__global__ void f32_to_bf16_kernel(const float* __restrict__ x, __bf16* __restrict__ y, size_t n) {
    for (size_t i = blockIdx.x * (size_t)blockDim.x + threadIdx.x; i < n;
         i += (size_t)gridDim.x * blockDim.x)
        y[i] = (__bf16)x[i];
}

// W [K,N] row-major f32  ->  Wt [N,Kpad] row-major bf16 (zero padded)
__global__ void transp_bf16_kernel(const float* __restrict__ W, __bf16* __restrict__ Wt,
                                   int K, int N, int Kpad) {
    size_t total = (size_t)N * Kpad;
    for (size_t idx = blockIdx.x * (size_t)blockDim.x + threadIdx.x; idx < total;
         idx += (size_t)gridDim.x * blockDim.x) {
        int n = (int)(idx / Kpad);
        int k = (int)(idx % Kpad);
        float v = (k < K) ? W[(size_t)k * N + n] : 0.f;
        Wt[idx] = (__bf16)v;
    }
}

// rel_w1 [656,512] -> WtRel [1024, KREL] bf16: cols 0..511 = s-half rows 0..327,
// cols 512..1023 = o-half rows 328..655
__global__ void relw_bf16_kernel(const float* __restrict__ relw1, __bf16* __restrict__ Wt) {
    size_t total = (size_t)1024 * KREL;
    for (size_t idx = blockIdx.x * (size_t)blockDim.x + threadIdx.x; idx < total;
         idx += (size_t)gridDim.x * blockDim.x) {
        int n = (int)(idx / KREL);
        int k = (int)(idx % KREL);
        float v = 0.f;
        if (k < 328) {
            if (n < HH) v = relw1[(size_t)k * HH + n];
            else        v = relw1[(size_t)(k + 328) * HH + (n - HH)];
        }
        Wt[idx] = (__bf16)v;
    }
}

// BN fold: S = g*rsqrt(v+eps); T = (b1 - m)*S + beta
__global__ void bn_st_kernel(const float* __restrict__ b1, const float* __restrict__ g,
                             const float* __restrict__ beta, const float* __restrict__ m,
                             const float* __restrict__ v, float* __restrict__ S,
                             float* __restrict__ T, int n) {
    int i = blockIdx.x * blockDim.x + threadIdx.x;
    if (i < n) {
        float s = g[i] * rsqrtf(v[i] + 1e-5f);
        S[i] = s;
        T[i] = (b1[i] - m[i]) * s + beta[i];
    }
}

// rel BN fold: relS[0..511]=relS[512..1023]=scale ; cshift = (b1 - m)*scale + beta
__global__ void rel_prep_kernel(const float* __restrict__ b1, const float* __restrict__ g,
                                const float* __restrict__ beta, const float* __restrict__ m,
                                const float* __restrict__ v, float* __restrict__ relS,
                                float* __restrict__ cshift) {
    int i = blockIdx.x * blockDim.x + threadIdx.x;
    if (i < HH) {
        float s = g[i] * rsqrtf(v[i] + 1e-5f);
        relS[i] = s;
        relS[HH + i] = s;
        cshift[i] = (b1[i] - m[i]) * s + beta[i];
    }
}

// ---------------------------------------------------------------------------
// pos MLP: featX[row, 0..127] = bf16(ReLU( ReLU(box@w1+b1) @ w2 + b2 ))
// one 128-thread block per row
// ---------------------------------------------------------------------------
__global__ void pos_feat_kernel(const float* __restrict__ box, const float* __restrict__ w1,
                                const float* __restrict__ b1, const float* __restrict__ w2,
                                const float* __restrict__ b2, __bf16* __restrict__ featX) {
    __shared__ float bx[9];
    __shared__ float h[GG];
    int row = blockIdx.x, t = threadIdx.x;
    if (t < 9) bx[t] = box[row * 9 + t];
    __syncthreads();
    float a = b1[t];
#pragma unroll
    for (int j = 0; j < 9; ++j) a += bx[j] * w1[j * GG + t];
    h[t] = a > 0.f ? a : 0.f;
    __syncthreads();
    float c = b2[t];
    for (int k = 0; k < GG; ++k) c += h[k] * w2[k * GG + t];
    featX[(size_t)row * KREL + t] = (__bf16)(c > 0.f ? c : 0.f);
}

// ---------------------------------------------------------------------------
// sem: featX[row, 128..327] = bf16(ReLU( softmax(logits) @ sem_w )); pad 328..383 = 0
// one 256-thread block per row
// ---------------------------------------------------------------------------
__global__ void sem_feat_kernel(const float* __restrict__ logits, const float* __restrict__ semw,
                                __bf16* __restrict__ featX) {
    __shared__ float p[CC];
    __shared__ float red[256];
    int row = blockIdx.x, t = threadIdx.x;
    float v = (t < CC) ? logits[(size_t)row * CC + t] : -3.4e38f;
    if (t < CC) p[t] = v;
    red[t] = v;
    __syncthreads();
    for (int s = 128; s > 0; s >>= 1) {
        if (t < s) red[t] = fmaxf(red[t], red[t + s]);
        __syncthreads();
    }
    float mx = red[0];
    __syncthreads();
    float e = (t < CC) ? expf(p[t] - mx) : 0.f;
    red[t] = e;
    __syncthreads();
    for (int s = 128; s > 0; s >>= 1) {
        if (t < s) red[t] += red[t + s];
        __syncthreads();
    }
    float inv = 1.f / red[0];
    __syncthreads();
    if (t < CC) p[t] = e * inv;
    __syncthreads();
    if (t < EE) {
        float c = 0.f;
        for (int k = 0; k < CC; ++k) c += p[k] * semw[k * EE + t];
        featX[(size_t)row * KREL + GG + t] = (__bf16)(c > 0.f ? c : 0.f);
    } else {
        // threads 200..255 zero pad columns 328..383
        featX[(size_t)row * KREL + 328 + (t - EE)] = (__bf16)0.f;
    }
}

// ---------------------------------------------------------------------------
// Generic bf16 WMMA GEMM:  C[M,N] = epilogue( A[M,KPAD] @ Bt[N,KPAD]^T )
//   epilogue: y = acc*S[c] + T[c]  (S==null -> 1, T==null -> 0), optional ReLU
//   OUTBF: 1 -> bf16 output Cb, 0 -> f32 output Cf
// Block = 256 threads (8 waves). Wave w computes rows [bx*16,+16) x cols
// [by*512 + w*64, +64) as 4 accumulators of v_wmma_f32_16x16x32_bf16.
// KPAD is a compile-time constant (multiple of 64) so all four B subtile
// addresses are immediate offsets off ONE base pointer pair (no per-tile
// 64-bit address registers -> no scratch spills). Explicit double buffering:
// fragments for k+32 are issued before the WMMAs consuming k.
// ---------------------------------------------------------------------------
template <int RELU, int OUTBF, int KPAD>
__global__ void wmma_gemm_kernel(const __bf16* __restrict__ A, const __bf16* __restrict__ Bt,
                                 int N,
                                 const float* __restrict__ S, const float* __restrict__ Tb,
                                 float* __restrict__ Cf, __bf16* __restrict__ Cb) {
    const int lane = threadIdx.x & 31;
    const int wave = threadIdx.x >> 5;
    const int m0 = blockIdx.x * 16;
    const int n0 = blockIdx.y * 512 + wave * 64;
    const int r  = lane & 15;
    const int kg = lane >> 4;

    // Per-lane base pointers (ISA fragment layouts):
    //  A (16-bit A layout): lane<16 -> K 0..7 & 16..23 ; lane>=16 -> +8
    //  B (16-bit B layout): lane<16 -> K 0..7 & 16..23 ; lane>=16 -> +8
    const __bf16* arow  = A + (size_t)(m0 + r) * KPAD + kg * 8;
    const __bf16* bbase = Bt + (size_t)(n0 + r) * KPAD + kg * 8;  // t-th subtile at +t*16*KPAD

    v8f acc[4];
#pragma unroll
    for (int t = 0; t < 4; ++t) acc[t] = (v8f){0.f, 0.f, 0.f, 0.f, 0.f, 0.f, 0.f, 0.f};

    Frag16 a0, a1, b0[4], b1[4];

    // prologue: stage k = 0
    {
        const uint4* ap = (const uint4*)(arow);
        a0.q[0] = ap[0];
        a0.q[1] = ap[2];
#pragma unroll
        for (int t = 0; t < 4; ++t) {
            const uint4* bp = (const uint4*)(bbase + t * 16 * KPAD);
            b0[t].q[0] = bp[0];
            b0[t].q[1] = bp[2];
        }
    }

    for (int k0 = 0; k0 < KPAD; k0 += 64) {
        // stage k0+32 while k0 fragments are in flight / being consumed
        {
            const uint4* ap = (const uint4*)(arow + k0 + 32);
            a1.q[0] = ap[0];
            a1.q[1] = ap[2];
#pragma unroll
            for (int t = 0; t < 4; ++t) {
                const uint4* bp = (const uint4*)(bbase + t * 16 * KPAD + k0 + 32);
                b1[t].q[0] = bp[0];
                b1[t].q[1] = bp[2];
            }
        }
        __builtin_prefetch(arow + k0 + 128, 0, 3);
#pragma unroll
        for (int t = 0; t < 4; ++t)
            acc[t] = __builtin_amdgcn_wmma_f32_16x16x32_bf16(
                false, a0.v, false, b0[t].v, (short)0, acc[t], false, false);

        // stage k0+64 (if any) while k0+32 fragments are consumed
        if (k0 + 64 < KPAD) {
            const uint4* ap = (const uint4*)(arow + k0 + 64);
            a0.q[0] = ap[0];
            a0.q[1] = ap[2];
#pragma unroll
            for (int t = 0; t < 4; ++t) {
                const uint4* bp = (const uint4*)(bbase + t * 16 * KPAD + k0 + 64);
                b0[t].q[0] = bp[0];
                b0[t].q[1] = bp[2];
            }
        }
#pragma unroll
        for (int t = 0; t < 4; ++t)
            acc[t] = __builtin_amdgcn_wmma_f32_16x16x32_bf16(
                false, a1.v, false, b1[t].v, (short)0, acc[t], false, false);
    }

#pragma unroll
    for (int t = 0; t < 4; ++t) {
        int cn = n0 + t * 16 + r;
        float s = S ? S[cn] : 1.0f;
        float b = Tb ? Tb[cn] : 0.0f;
        int mb = m0 + kg * 8;
#pragma unroll
        for (int i = 0; i < 8; ++i) {
            float y = acc[t][i] * s + b;
            if (RELU) y = y > 0.f ? y : 0.f;
            size_t idx = (size_t)(mb + i) * N + cn;
            if (OUTBF) Cb[idx] = (__bf16)y;
            else       Cf[idx] = y;
        }
    }
}

// ---------------------------------------------------------------------------
// Per-pair kernel: one wave32 per pair.
// relness = sum_c ReLU(Ls'[s,c] + Lo'[o,c] + cshift[c]) * w2[c]
//         + sum_c sf[s,c]*of[o,c] + rel_b2
// out[s*NN + o] = sigmoid(relness)
// ---------------------------------------------------------------------------
__global__ void pair_kernel(const int* __restrict__ pairs, const float* __restrict__ LsLo,
                            const float* __restrict__ sf, const float* __restrict__ of,
                            const float* __restrict__ cshift, const float* __restrict__ w2,
                            const float* __restrict__ b2, float* __restrict__ out, int P) {
    int gw = (int)((blockIdx.x * (size_t)blockDim.x + threadIdx.x) >> 5);
    int lane = threadIdx.x & 31;
    if (gw >= P) return;
    int s = pairs[2 * gw];
    int o = pairs[2 * gw + 1];
    const float4* ls = (const float4*)(LsLo + (size_t)s * 1024);
    const float4* lo = (const float4*)(LsLo + (size_t)o * 1024 + HH);
    const float4* xs = (const float4*)(sf + (size_t)s * HH);
    const float4* yo = (const float4*)(of + (size_t)o * HH);
    const float4* cs = (const float4*)cshift;
    const float4* wv = (const float4*)w2;
    float acc = 0.f;
#pragma unroll
    for (int i = 0; i < 4; ++i) {
        int c = i * 32 + lane;
        float4 a = ls[c], b = lo[c], x = xs[c], y = yo[c], h = cs[c], w = wv[c];
        float t0 = a.x + b.x + h.x; t0 = t0 > 0.f ? t0 : 0.f;
        float t1 = a.y + b.y + h.y; t1 = t1 > 0.f ? t1 : 0.f;
        float t2 = a.z + b.z + h.z; t2 = t2 > 0.f ? t2 : 0.f;
        float t3 = a.w + b.w + h.w; t3 = t3 > 0.f ? t3 : 0.f;
        acc += t0 * w.x + t1 * w.y + t2 * w.z + t3 * w.w;
        acc += x.x * y.x + x.y * y.y + x.z * y.z + x.w * y.w;
    }
    for (int off = 16; off > 0; off >>= 1) acc += __shfl_xor(acc, off, 32);
    if (lane == 0) {
        float r = acc + b2[0];
        out[(size_t)s * NN + o] = 1.f / (1.f + expf(-r));
    }
}

// ---------------------------------------------------------------------------
// Host launcher
// ---------------------------------------------------------------------------
extern "C" void kernel_launch(void* const* d_in, const int* in_sizes, int n_in,
                              void* d_out, int out_size, void* d_ws, size_t ws_size,
                              hipStream_t stream) {
    const float* box     = (const float*)d_in[0];
    const float* logits  = (const float*)d_in[1];
    const float* roi     = (const float*)d_in[2];
    const int*   pairs   = (const int*)d_in[3];
    const float* pos_w1  = (const float*)d_in[4];
    const float* pos_b1  = (const float*)d_in[5];
    const float* pos_w2  = (const float*)d_in[6];
    const float* pos_b2  = (const float*)d_in[7];
    const float* sem_w   = (const float*)d_in[8];
    const float* rel_w1  = (const float*)d_in[9];
    const float* rel_b1  = (const float*)d_in[10];
    const float* rel_g   = (const float*)d_in[11];
    const float* rel_be  = (const float*)d_in[12];
    const float* rel_m   = (const float*)d_in[13];
    const float* rel_v   = (const float*)d_in[14];
    const float* rel_w2  = (const float*)d_in[15];
    const float* rel_b2  = (const float*)d_in[16];
    const float* sub_w1  = (const float*)d_in[17];
    const float* sub_b1  = (const float*)d_in[18];
    const float* sub_g   = (const float*)d_in[19];
    const float* sub_be  = (const float*)d_in[20];
    const float* sub_m   = (const float*)d_in[21];
    const float* sub_v   = (const float*)d_in[22];
    const float* sub_w2  = (const float*)d_in[23];
    const float* sub_b2  = (const float*)d_in[24];
    const float* obj_w1  = (const float*)d_in[25];
    const float* obj_b1  = (const float*)d_in[26];
    const float* obj_g   = (const float*)d_in[27];
    const float* obj_be  = (const float*)d_in[28];
    const float* obj_m   = (const float*)d_in[29];
    const float* obj_v   = (const float*)d_in[30];
    const float* obj_w2  = (const float*)d_in[31];
    const float* obj_b2  = (const float*)d_in[32];
    // satt: wq(33) bq(34) wk(35) bk(36) wv(37) bv(38) wo(39) bo(40)  -- wq/wk dead
    const float* satt_wv = (const float*)d_in[37];
    const float* satt_bv = (const float*)d_in[38];
    const float* satt_wo = (const float*)d_in[39];
    const float* satt_bo = (const float*)d_in[40];
    const float* oatt_wv = (const float*)d_in[45];
    const float* oatt_bv = (const float*)d_in[46];
    const float* oatt_wo = (const float*)d_in[47];
    const float* oatt_bo = (const float*)d_in[48];

    const int P = in_sizes[3] / 2;

    // ---- workspace bump allocator (256B aligned) ----
    char* base = (char*)d_ws;
    size_t off = 0;
    auto alloc = [&](size_t bytes) -> void* {
        void* p = base + off;
        off = (off + bytes + 255) & ~(size_t)255;
        return p;
    };
    __bf16* roiB    = (__bf16*)alloc((size_t)NN * DROI * 2);
    __bf16* WtSub1  = (__bf16*)alloc((size_t)HH * DROI * 2);
    __bf16* WtObj1  = (__bf16*)alloc((size_t)HH * DROI * 2);
    __bf16* WtSub2  = (__bf16*)alloc((size_t)HH * HH * 2);
    __bf16* WtSubV  = (__bf16*)alloc((size_t)HH * HH * 2);
    __bf16* WtSubO  = (__bf16*)alloc((size_t)HH * HH * 2);
    __bf16* WtObj2  = (__bf16*)alloc((size_t)HH * HH * 2);
    __bf16* WtObjV  = (__bf16*)alloc((size_t)HH * HH * 2);
    __bf16* WtObjO  = (__bf16*)alloc((size_t)HH * HH * 2);
    __bf16* WtRel   = (__bf16*)alloc((size_t)1024 * KREL * 2);
    __bf16* featX   = (__bf16*)alloc((size_t)NN * KREL * 2);
    __bf16* Tsub    = (__bf16*)alloc((size_t)NN * HH * 2);
    __bf16* Tobj    = (__bf16*)alloc((size_t)NN * HH * 2);
    __bf16* Xa      = (__bf16*)alloc((size_t)NN * HH * 2);
    __bf16* Xb      = (__bf16*)alloc((size_t)NN * HH * 2);
    float*  sfF     = (float*)alloc((size_t)NN * HH * 4);
    float*  ofF     = (float*)alloc((size_t)NN * HH * 4);
    float*  LsLo    = (float*)alloc((size_t)NN * 1024 * 4);
    float*  Ssub1   = (float*)alloc(HH * 4);
    float*  Tsub1   = (float*)alloc(HH * 4);
    float*  Sobj1   = (float*)alloc(HH * 4);
    float*  Tobj1   = (float*)alloc(HH * 4);
    float*  relS    = (float*)alloc(1024 * 4);
    float*  cshift  = (float*)alloc(HH * 4);

    float* out = (float*)d_out;

    // ---- prep ----
    zero_f32_kernel<<<2048, 256, 0, stream>>>(out, (size_t)NN * NN);
    f32_to_bf16_kernel<<<4096, 256, 0, stream>>>(roi, roiB, (size_t)NN * DROI);
    transp_bf16_kernel<<<4096, 256, 0, stream>>>(sub_w1, WtSub1, DROI, HH, DROI);
    transp_bf16_kernel<<<4096, 256, 0, stream>>>(obj_w1, WtObj1, DROI, HH, DROI);
    transp_bf16_kernel<<<1024, 256, 0, stream>>>(sub_w2, WtSub2, HH, HH, HH);
    transp_bf16_kernel<<<1024, 256, 0, stream>>>(satt_wv, WtSubV, HH, HH, HH);
    transp_bf16_kernel<<<1024, 256, 0, stream>>>(satt_wo, WtSubO, HH, HH, HH);
    transp_bf16_kernel<<<1024, 256, 0, stream>>>(obj_w2, WtObj2, HH, HH, HH);
    transp_bf16_kernel<<<1024, 256, 0, stream>>>(oatt_wv, WtObjV, HH, HH, HH);
    transp_bf16_kernel<<<1024, 256, 0, stream>>>(oatt_wo, WtObjO, HH, HH, HH);
    relw_bf16_kernel<<<1024, 256, 0, stream>>>(rel_w1, WtRel);
    bn_st_kernel<<<2, 256, 0, stream>>>(sub_b1, sub_g, sub_be, sub_m, sub_v, Ssub1, Tsub1, HH);
    bn_st_kernel<<<2, 256, 0, stream>>>(obj_b1, obj_g, obj_be, obj_m, obj_v, Sobj1, Tobj1, HH);
    rel_prep_kernel<<<2, 256, 0, stream>>>(rel_b1, rel_g, rel_be, rel_m, rel_v, relS, cshift);
    pos_feat_kernel<<<NN, GG, 0, stream>>>(box, pos_w1, pos_b1, pos_w2, pos_b2, featX);
    sem_feat_kernel<<<NN, 256, 0, stream>>>(logits, sem_w, featX);

    // ---- GEMMs (bf16 WMMA, f32 accumulate) ----
    dim3 blk(256);
    dim3 g512(NN / 16, 1);   // N = 512
    dim3 g1024(NN / 16, 2);  // N = 1024

    // sub path: roi@W1 (+BN fold, ReLU) -> @w2 -> @wv -> @wo
    wmma_gemm_kernel<1, 1, DROI><<<g512, blk, 0, stream>>>(roiB, WtSub1, HH, Ssub1, Tsub1, nullptr, Tsub);
    wmma_gemm_kernel<0, 1, HH><<<g512, blk, 0, stream>>>(Tsub, WtSub2, HH, nullptr, sub_b2, nullptr, Xa);
    wmma_gemm_kernel<0, 1, HH><<<g512, blk, 0, stream>>>(Xa, WtSubV, HH, nullptr, satt_bv, nullptr, Xb);
    wmma_gemm_kernel<0, 0, HH><<<g512, blk, 0, stream>>>(Xb, WtSubO, HH, nullptr, satt_bo, sfF, nullptr);
    // obj path
    wmma_gemm_kernel<1, 1, DROI><<<g512, blk, 0, stream>>>(roiB, WtObj1, HH, Sobj1, Tobj1, nullptr, Tobj);
    wmma_gemm_kernel<0, 1, HH><<<g512, blk, 0, stream>>>(Tobj, WtObj2, HH, nullptr, obj_b2, nullptr, Xa);
    wmma_gemm_kernel<0, 1, HH><<<g512, blk, 0, stream>>>(Xa, WtObjV, HH, nullptr, oatt_bv, nullptr, Xb);
    wmma_gemm_kernel<0, 0, HH><<<g512, blk, 0, stream>>>(Xb, WtObjO, HH, nullptr, oatt_bo, ofF, nullptr);
    // relness node-level GEMM: featX @ [W_s | W_o], scaled by folded BN
    wmma_gemm_kernel<0, 0, KREL><<<g1024, blk, 0, stream>>>(featX, WtRel, 1024, relS, nullptr, LsLo, nullptr);

    // ---- per-pair epilogue + scatter ----
    int pairBlocks = (P * 32 + 255) / 256;
    pair_kernel<<<pairBlocks, 256, 0, stream>>>(pairs, LsLo, sfF, ofF, cshift, rel_w2, rel_b2, out, P);
}
